// RNN_89137751261566
// MI455X (gfx1250) — matrix operand order repaired and checked by reading, not compiled, not run on hardware
//
#include <hip/hip_runtime.h>
#include <math.h>

#define IN_DIM   256
#define LATENT   1024
#define OUT_DIM  256
#define BATCH    128
#define T_STEPS  1024
#define HROW     1032   // padded LDS row stride (bf16 elems): 2064B, bank-spread

typedef __bf16 bf16_t;
typedef __attribute__((ext_vector_type(16))) __bf16         v16bf;
typedef __attribute__((ext_vector_type(8)))  __bf16         v8bf;
typedef __attribute__((ext_vector_type(8)))  float          v8f;
typedef __attribute__((ext_vector_type(8)))  unsigned short u16x8;
typedef __attribute__((ext_vector_type(4)))  float          f32x4;

// ---------------------------------------------------------------------------
// WMMA + fragment helpers (CDNA5 wave32, v_wmma_f32_16x16x32_bf16)
// ---------------------------------------------------------------------------
__device__ __forceinline__ v8f wmma_bf16(v16bf a, v16bf b, v8f c) {
  return __builtin_amdgcn_wmma_f32_16x16x32_bf16(false, a, false, b,
                                                 (short)0, c, false, false);
}

// A 16x32 bf16 fragment (row-major source, global OR LDS pointer):
// lane group g: elems 0..7 <- K = g*8.., elems 8..15 <- K = 16+g*8..
__device__ __forceinline__ v16bf load_a_bf(const bf16_t* base, int g) {
  union { u16x8 u[2]; v16bf v; } c;
  c.u[0] = *(const u16x8*)(base + g * 8);
  c.u[1] = *(const u16x8*)(base + 16 + g * 8);
  return c.v;
}

// Same A fragment sourced from f32 memory with in-register cvt to bf16.
__device__ __forceinline__ v16bf load_a_f32(const float* __restrict__ base, int g) {
  f32x4 a0 = *(const f32x4*)(base + g * 8);
  f32x4 a1 = *(const f32x4*)(base + g * 8 + 4);
  f32x4 b0 = *(const f32x4*)(base + 16 + g * 8);
  f32x4 b1 = *(const f32x4*)(base + 16 + g * 8 + 4);
  v16bf r;
#pragma unroll
  for (int i = 0; i < 4; ++i) {
    r[i]      = (bf16_t)a0[i];
    r[4 + i]  = (bf16_t)a1[i];
    r[8 + i]  = (bf16_t)b0[i];
    r[12 + i] = (bf16_t)b1[i];
  }
  return r;
}

// B 32x16 bf16 fragment: lane n = lane&15 selects a row of W (= column of B),
// lane group g: elems 0..7 <- K = g*8.., elems 8..15 <- K = 16+g*8.. (matches A).
__device__ __forceinline__ v16bf load_b(const bf16_t* __restrict__ wrow, int g) {
  union { u16x8 u[2]; v16bf v; } c;
  c.u[0] = *(const u16x8*)(wrow + g * 8);
  c.u[1] = *(const u16x8*)(wrow + 16 + g * 8);
  return c.v;
}

// Fast tanh: v_exp_f32 + v_rcp_f32 (~6 VALU ops vs ~40 for libm tanhf).
__device__ __forceinline__ float fast_tanh(float v) {
  float z = fminf(fmaxf(v, -9.0f), 9.0f);
  float e = __builtin_amdgcn_exp2f(z * 2.8853900817779268f);
  return (e - 1.0f) * __builtin_amdgcn_rcpf(e + 1.0f);
}

// ---------------------------------------------------------------------------
// Prep: f32 -> bf16 weight conversion
// ---------------------------------------------------------------------------
__global__ void rnn_prep_kernel(const float* __restrict__ W_lin,
                                const float* __restrict__ W_dec,
                                bf16_t* __restrict__ Wx, bf16_t* __restrict__ Wh,
                                bf16_t* __restrict__ Wd) {
  const int stride = gridDim.x * blockDim.x;
  const int i0 = blockIdx.x * blockDim.x + threadIdx.x;
  for (int i = i0; i < LATENT * (IN_DIM + LATENT); i += stride) {
    int r = i / (IN_DIM + LATENT);
    int c = i - r * (IN_DIM + LATENT);
    bf16_t v = (bf16_t)W_lin[i];
    if (c < IN_DIM) Wx[r * IN_DIM + c] = v;
    else            Wh[r * LATENT + (c - IN_DIM)] = v;
  }
  for (int i = i0; i < OUT_DIM * LATENT; i += stride) Wd[i] = (bf16_t)W_dec[i];
}

// ---------------------------------------------------------------------------
// x f32 -> bf16, one pass (8 elems/thread, vectorized)
// ---------------------------------------------------------------------------
__global__ __launch_bounds__(256)
void rnn_xcvt_kernel(const float* __restrict__ x, bf16_t* __restrict__ XB) {
  const size_t i = ((size_t)blockIdx.x * blockDim.x + threadIdx.x) * 8;
  f32x4 a = *(const f32x4*)(x + i);
  f32x4 b = *(const f32x4*)(x + i + 4);
  v8bf o;
#pragma unroll
  for (int j = 0; j < 4; ++j) {
    o[j]     = (bf16_t)a[j];
    o[4 + j] = (bf16_t)b[j];
  }
  *(v8bf*)(XB + i) = o;
}

// ---------------------------------------------------------------------------
// Parallel x-projection: XP[t*B+b][n] = sum_k x[b][t][k] * Wx[n][k]  (bf16)
// M = T*B = 131072, N = 1024, K = 256. One wave -> 16x64 strip (4 acc chains),
// 2-stage software pipeline.
// ---------------------------------------------------------------------------
__global__ __launch_bounds__(256)
void rnn_xproj_kernel(const bf16_t* __restrict__ XB, const bf16_t* __restrict__ Wx,
                      bf16_t* __restrict__ XP) {
  const int lane = threadIdx.x & 31;
  const int wave = blockIdx.x * 8 + (threadIdx.x >> 5);  // 0..131071
  const int mtile = wave >> 4;   // 0..8191
  const int ngrp  = wave & 15;   // 0..15 -> n-tiles ngrp*4 .. +3
  const int lid = lane & 15;
  const int g   = lane >> 4;

  const int rowM = mtile * 16 + lid;           // flat XP row = t*128 + b
  const int b = rowM & (BATCH - 1);
  const int t = rowM >> 7;
  const bf16_t* xrow = XB + ((size_t)b * T_STEPS + t) * IN_DIM;

  const int nb = ngrp * 64 + lid;
  const bf16_t* w0 = Wx + (size_t)(nb)      * IN_DIM;
  const bf16_t* w1 = Wx + (size_t)(nb + 16) * IN_DIM;
  const bf16_t* w2 = Wx + (size_t)(nb + 32) * IN_DIM;
  const bf16_t* w3 = Wx + (size_t)(nb + 48) * IN_DIM;

  v8f a0 = {}, a1 = {}, a2 = {}, a3 = {};
  v16bf ac  = load_a_bf(xrow, g);
  v16bf b0c = load_b(w0, g), b1c = load_b(w1, g);
  v16bf b2c = load_b(w2, g), b3c = load_b(w3, g);
#pragma unroll
  for (int kk = 0; kk < IN_DIM; kk += 32) {
    v16bf an = ac, b0n = b0c, b1n = b1c, b2n = b2c, b3n = b3c;
    if (kk + 32 < IN_DIM) {           // issue next-iter loads before this WMMA
      an  = load_a_bf(xrow + kk + 32, g);
      b0n = load_b(w0 + kk + 32, g);
      b1n = load_b(w1 + kk + 32, g);
      b2n = load_b(w2 + kk + 32, g);
      b3n = load_b(w3 + kk + 32, g);
    }
    a0 = wmma_bf16(ac, b0c, a0);
    a1 = wmma_bf16(ac, b1c, a1);
    a2 = wmma_bf16(ac, b2c, a2);
    a3 = wmma_bf16(ac, b3c, a3);
    ac = an; b0c = b0n; b1c = b1n; b2c = b2n; b3c = b3n;
  }
  const int orow = mtile * 16 + g * 8;
#pragma unroll
  for (int e = 0; e < 8; ++e) {
    size_t base = (size_t)(orow + e) * LATENT + ngrp * 64 + lid;
    XP[base]      = (bf16_t)a0[e];
    XP[base + 16] = (bf16_t)a1[e];
    XP[base + 32] = (bf16_t)a2[e];
    XP[base + 48] = (bf16_t)a3[e];
  }
}

// ---------------------------------------------------------------------------
// Sequential recurrence. One workgroup (32 waves) owns one 16-row batch strip;
// hidden state lives in double-buffered LDS; one __syncthreads per step.
// XMODE: 0 = x fused (f32 src), 1 = x fused (bf16 src), 2 = XP precomputed.
// ---------------------------------------------------------------------------
template <int XMODE>
__global__ __launch_bounds__(1024, 1)
void rnn_seq_kernel(const float* __restrict__ x, const bf16_t* __restrict__ XB,
                    const bf16_t* __restrict__ XP,
                    const bf16_t* __restrict__ Wx, const bf16_t* __restrict__ Wh,
                    bf16_t* __restrict__ Hfinal) {
  __shared__ bf16_t Hs[2][16 * HROW];   // 2 x 33 KB hidden-state strip

  const int lane = threadIdx.x & 31;
  const int wave = threadIdx.x >> 5;     // 0..31
  const int lid = lane & 15;
  const int g   = lane >> 4;
  const int gm  = blockIdx.x * 16;       // global batch-row base of this strip

  const int n0 = (2 * wave) * 16 + lid;  // W row / output col, tile 0
  const int n1 = n0 + 16;                // tile 1
  const bf16_t* wh0 = Wh + (size_t)n0 * LATENT;
  const bf16_t* wh1 = Wh + (size_t)n1 * LATENT;
  const bf16_t* wx0 = Wx + (size_t)n0 * IN_DIM;
  const bf16_t* wx1 = Wx + (size_t)n1 * IN_DIM;
  const float*  xrow_f = x  + (size_t)(gm + lid) * T_STEPS * IN_DIM;
  const bf16_t* xrow_b = XB + (size_t)(gm + lid) * T_STEPS * IN_DIM;

  const int om = g * 8;                  // local output-row base

  // h0 = 0
  for (int i = threadIdx.x; i < 16 * HROW; i += 1024) Hs[0][i] = (bf16_t)0.0f;
  __syncthreads();

  for (int t = 0; t < T_STEPS; ++t) {
    const int rb = t & 1;                      // read buffer
    const int wb = rb ^ 1;                     // write buffer
    const bf16_t* hrow = &Hs[rb][lid * HROW];  // A-fragment source row (LDS)

    v8f acc0 = {};
    v8f acc1 = {};
    float xv0[8], xv1[8];

    if (XMODE == 2) {
      // Precomputed x-projection tile (bf16, streams HBM; consumed post-loop).
      const bf16_t* xp = XP + ((size_t)t * BATCH + gm) * LATENT;
      if (t + 1 < T_STEPS)
        __builtin_prefetch(xp + (size_t)BATCH * LATENT + threadIdx.x * 32, 0, 0);
#pragma unroll
      for (int e = 0; e < 8; ++e) {
        xv0[e] = (float)xp[(om + e) * LATENT + n0];
        xv1[e] = (float)xp[(om + e) * LATENT + n1];
      }
    } else {
      // Fused x_t @ Wx^T (K += 256).
#pragma unroll
      for (int kk = 0; kk < IN_DIM; kk += 32) {
        v16bf a = (XMODE == 1) ? load_a_bf(xrow_b + (size_t)t * IN_DIM + kk, g)
                               : load_a_f32(xrow_f + (size_t)t * IN_DIM + kk, g);
        acc0 = wmma_bf16(a, load_b(wx0 + kk, g), acc0);
        acc1 = wmma_bf16(a, load_b(wx1 + kk, g), acc1);
      }
    }

    // Recurrent GEMM: A from LDS (ds_load_b128), B = Wh rows from L2.
    // 2-stage software pipeline: next-iter fragments issued before WMMAs.
    v16bf ac  = load_a_bf(hrow, g);
    v16bf b0c = load_b(wh0, g);
    v16bf b1c = load_b(wh1, g);
#pragma unroll 4
    for (int kk = 0; kk < LATENT; kk += 32) {
      v16bf an = ac, b0n = b0c, b1n = b1c;
      if (kk + 32 < LATENT) {
        an  = load_a_bf(hrow + kk + 32, g);
        b0n = load_b(wh0 + kk + 32, g);
        b1n = load_b(wh1 + kk + 32, g);
      }
      acc0 = wmma_bf16(ac, b0c, acc0);
      acc1 = wmma_bf16(ac, b1c, acc1);
      ac = an; b0c = b0n; b1c = b1n;
    }

    if (t + 1 < T_STEPS) {
      // Write next state into the other buffer (no barrier needed before).
#pragma unroll
      for (int e = 0; e < 8; ++e) {
        float h0 = fast_tanh(acc0[e] + (XMODE == 2 ? xv0[e] : 0.0f));
        float h1 = fast_tanh(acc1[e] + (XMODE == 2 ? xv1[e] : 0.0f));
        Hs[wb][(om + e) * HROW + n0] = (bf16_t)h0;
        Hs[wb][(om + e) * HROW + n1] = (bf16_t)h1;
      }
    } else {
      // Last step: ship h_T straight from registers to global for decode.
#pragma unroll
      for (int e = 0; e < 8; ++e) {
        float h0 = fast_tanh(acc0[e] + (XMODE == 2 ? xv0[e] : 0.0f));
        float h1 = fast_tanh(acc1[e] + (XMODE == 2 ? xv1[e] : 0.0f));
        Hfinal[(size_t)(gm + om + e) * LATENT + n0] = (bf16_t)h0;
        Hfinal[(size_t)(gm + om + e) * LATENT + n1] = (bf16_t)h1;
      }
    }
    __syncthreads();   // writes to Hs[wb] visible before step t+1 reads it
  }
}

// ---------------------------------------------------------------------------
// Decode: out = tanh(h_final @ W_dec^T + b_dec), M=128 N=256 K=1024
// ---------------------------------------------------------------------------
__global__ __launch_bounds__(128)
void rnn_decode_kernel(const bf16_t* __restrict__ H,
                       const bf16_t* __restrict__ Wd,
                       const float* __restrict__ b_dec,
                       float* __restrict__ out) {
  const int lane = threadIdx.x & 31;
  const int wave = blockIdx.x * (blockDim.x >> 5) + (threadIdx.x >> 5); // 0..127
  const int tile_m = wave >> 4;
  const int tile_n = wave & 15;
  const int lid = lane & 15;
  const int g   = lane >> 4;

  const bf16_t* hrow = H  + (size_t)(tile_m * 16 + lid) * LATENT;
  const bf16_t* wrow = Wd + (size_t)(tile_n * 16 + lid) * LATENT;

  v8f acc = {};
  v16bf ac = load_a_bf(hrow, g);
  v16bf bc = load_b(wrow, g);
#pragma unroll 8
  for (int kk = 0; kk < LATENT; kk += 32) {
    v16bf an = ac, bn = bc;
    if (kk + 32 < LATENT) {
      an = load_a_bf(hrow + kk + 32, g);
      bn = load_b(wrow + kk + 32, g);
    }
    acc = wmma_bf16(ac, bc, acc);
    ac = an; bc = bn;
  }

  const int om = tile_m * 16 + g * 8;
  const int on = tile_n * 16 + lid;
  const float bias = b_dec[on];
#pragma unroll
  for (int e = 0; e < 8; ++e)
    out[(size_t)(om + e) * OUT_DIM + on] = fast_tanh(acc[e] + bias);
}

// ---------------------------------------------------------------------------
// Launch
// ---------------------------------------------------------------------------
extern "C" void kernel_launch(void* const* d_in, const int* in_sizes, int n_in,
                              void* d_out, int out_size, void* d_ws, size_t ws_size,
                              hipStream_t stream) {
  const float* x     = (const float*)d_in[0];  // (B, T, IN_DIM)
  const float* W_lin = (const float*)d_in[1];  // (LATENT, IN_DIM+LATENT)
  const float* W_dec = (const float*)d_in[2];  // (OUT_DIM, LATENT)
  const float* b_dec = (const float*)d_in[3];  // (OUT_DIM,)
  float* out = (float*)d_out;                  // (B, OUT_DIM) f32

  char* ws = (char*)d_ws;
  bf16_t* Wh     = (bf16_t*)(ws);                                         // 2 MB
  bf16_t* Wx     = (bf16_t*)(ws + (size_t)2 * 1024 * 1024);               // 512 KB
  bf16_t* Wd     = (bf16_t*)(ws + (size_t)2 * 1024 * 1024 + 512 * 1024);  // 512 KB
  bf16_t* Hfinal = (bf16_t*)(ws + (size_t)3 * 1024 * 1024);               // 256 KB

  const size_t XB_OFF   = (size_t)4 * 1024 * 1024;
  const size_t XB_BYTES = (size_t)BATCH * T_STEPS * IN_DIM * sizeof(bf16_t);  // 64 MB
  const size_t XP_OFF   = XB_OFF + XB_BYTES;
  const size_t XP_BYTES = (size_t)T_STEPS * BATCH * LATENT * sizeof(bf16_t);  // 256 MB
  bf16_t* XB = (bf16_t*)(ws + XB_OFF);
  bf16_t* XP = (bf16_t*)(ws + XP_OFF);

  const bool have_xb = (ws_size >= XB_OFF + XB_BYTES);            // 68 MB
  const bool have_xp = (ws_size >= XP_OFF + XP_BYTES);            // 324 MB

  rnn_prep_kernel<<<1024, 256, 0, stream>>>(W_lin, W_dec, Wx, Wh, Wd);
  if (have_xb) {
    const int n8 = BATCH * T_STEPS * IN_DIM / 8;                  // 4,194,304
    rnn_xcvt_kernel<<<n8 / 256, 256, 0, stream>>>(x, XB);
  }
  if (have_xp) {
    rnn_xproj_kernel<<<16384, 256, 0, stream>>>(XB, Wx, XP);
    rnn_seq_kernel<2><<<8, 1024, 0, stream>>>(x, XB, XP, Wx, Wh, Hfinal);
  } else if (have_xb) {
    rnn_seq_kernel<1><<<8, 1024, 0, stream>>>(x, XB, nullptr, Wx, Wh, Hfinal);
  } else {
    rnn_seq_kernel<0><<<8, 1024, 0, stream>>>(x, nullptr, nullptr, Wx, Wh, Hfinal);
  }
  rnn_decode_kernel<<<32, 128, 0, stream>>>(Hfinal, Wd, b_dec, out);
}
